// CrossCorrelation_40870908788870
// MI455X (gfx1250) — matrix operand order repaired and checked by reading, not compiled, not run on hardware
//
#include <hip/hip_runtime.h>

// ---------------------------------------------------------------------------
// CDNA5 / gfx1250 circular cross-correlation via bf16 WMMA (hi/lo split).
//
// out[b,p(i<=j),sy,sx] = sum_{y,x} xs_i[(y+dy)%128,(x+dx)%128] * xs_j[y,x]
//
// Block = (b, 7 consecutive sy, sx in {g,g+8,g+16}).  A rows for adjacent sy
// form a sliding window: per y only ONE new A row + ONE B row enter LDS,
// serving up to 21 (sy,sx) accumulators.  8-slot circular A window +
// double-buffered B row, one barrier per y.  Shift count templated (branch-
// free inner loop).
// ---------------------------------------------------------------------------

typedef __attribute__((ext_vector_type(16))) __bf16 v16bf;
typedef __attribute__((ext_vector_type(8)))  float  v8f;

#define BATCH  8
#define CHANS  32
#define HW     128
#define NPIX   (HW * HW)          // 16384
#define NPAIR  528                // 32*33/2
#define NS     21                 // shifts per axis
#define NGX    8                  // sx groups {g, g+8, g+16}
#define NSYG   3                  // sy groups of 7
#define STRIDE 144                // LDS row stride (bf16): 128 + 16 dup window
#define NTHR   672                // 21 waves

#define AW_PLANE_E (CHANS * STRIDE)        // 4608 elems
#define AW_SLOT_E  (2 * AW_PLANE_E)        // hi+lo planes
#define AW_ELEMS   (8 * AW_SLOT_E)         // 8 circular slots
#define BB_BUF_E   (2 * AW_PLANE_E)
#define SMEM_BYTES ((AW_ELEMS + 2 * BB_BUF_E) * 2)   // 184320 B

// ---- bf16 helpers (bit-level) ---------------------------------------------
__device__ __forceinline__ unsigned bf_bits(float f) {           // RNE f32->bf16
    unsigned u = __builtin_bit_cast(unsigned, f);
    return (u + 0x7FFFu + ((u >> 16) & 1u)) >> 16;
}
__device__ __forceinline__ float bfbits_to_f(unsigned b) {
    return __builtin_bit_cast(float, b << 16);
}
__device__ __forceinline__ __bf16 u16_to_bf(unsigned u) {
    return __builtin_bit_cast(__bf16, (unsigned short)(u & 0xFFFFu));
}
__device__ __forceinline__ void unpack4(uint4 p, uint2& hv, uint2& lv) {
    hv.x = (p.x & 0xFFFFu) | (p.y << 16);
    hv.y = (p.z & 0xFFFFu) | (p.w << 16);
    lv.x = (p.x >> 16) | (p.y & 0xFFFF0000u);
    lv.y = (p.z >> 16) | (p.w & 0xFFFF0000u);
}

// ---- fragment loads (two 16B ds_load_b128 each) ---------------------------
__device__ __forceinline__ v16bf ld_frag(const __bf16* p0, const __bf16* p1) {
    union { uint4 q[2]; v16bf v; } u;
    u.q[0] = *(const uint4*)p0;
    u.q[1] = *(const uint4*)p1;
    return u.v;
}
// A 16x32 bf16 (ISA 7.12.2 A table): lane<16: M=lane, K=kb+{0..7,16..23};
// lane>=16: M=lane-16, K=kb+{8..15,24..31}.  sof = shift offset (0/8/16).
__device__ __forceinline__ v16bf ldA(const __bf16* base, int mb, int kb, int sof, int lane) {
    const int m  = mb + (lane & 15);
    const int kk = kb + sof + ((lane >> 4) << 3);
    const __bf16* p = base + m * STRIDE + kk;
    return ld_frag(p, p + 16);
}
// B 32x16 bf16 (row striped across lanes, like C/D): lane=N,
// lane<16: K=kb+{0..7,16..23}; lane>=16: K=kb+{8..15,24..31} (matches A).
__device__ __forceinline__ v16bf ldB(const __bf16* base, int nb, int kb, int lane) {
    const int n  = nb + (lane & 15);
    const int kk = kb + ((lane >> 4) << 3);
    const __bf16* p = base + n * STRIDE + kk;
    return ld_frag(p, p + 16);
}

__device__ __forceinline__ v8f wmma_bf16(v16bf a, v16bf b, v8f c) {
    return __builtin_amdgcn_wmma_f32_16x16x32_bf16(
        false, a, false, b, (short)0, c, false, false);
}

// ---- LDS row fills --------------------------------------------------------
// A row: SOURCE-ALIGNED uint4 global load; rotated scatter (b16 stores, no
// alignment requirement); first 16 elems duplicated at t+128 (wrap window).
__device__ __forceinline__ void fill_a_row(__bf16* aw, const unsigned* __restrict__ base,
                                           int rowi, int dx0, int tid) {
    __bf16* d0 = aw;
    __bf16* d1 = aw + AW_PLANE_E;
    for (int idx = tid; idx < CHANS * (HW / 4); idx += NTHR) {
        const int c  = idx >> 5;
        const int x0 = (idx & 31) << 2;
        const uint4 p = *(const uint4*)&base[c * NPIX + rowi * HW + x0];
        const unsigned px[4] = {p.x, p.y, p.z, p.w};
        #pragma unroll
        for (int e = 0; e < 4; ++e) {
            const int t = (x0 + e - dx0) & 127;
            const __bf16 h = u16_to_bf(px[e]);
            const __bf16 l = u16_to_bf(px[e] >> 16);
            d0[c * STRIDE + t] = h;
            d1[c * STRIDE + t] = l;
            if (t < STRIDE - HW) {                 // dup wrap window
                d0[c * STRIDE + t + HW] = h;
                d1[c * STRIDE + t + HW] = l;
            }
        }
    }
}
// B row: fully aligned both sides -> uint4 load, uint2 (b64) stores.
__device__ __forceinline__ void fill_b_row(__bf16* bb, const unsigned* __restrict__ base,
                                           int rowi, int tid) {
    __bf16* d0 = bb;
    __bf16* d1 = bb + AW_PLANE_E;
    for (int idx = tid; idx < CHANS * (HW / 4); idx += NTHR) {
        const int c  = idx >> 5;
        const int t0 = (idx & 31) << 2;
        const uint4 p = *(const uint4*)&base[c * NPIX + rowi * HW + t0];
        uint2 hv, lv; unpack4(p, hv, lv);
        *(uint2*)&d0[c * STRIDE + t0] = hv;
        *(uint2*)&d1[c * STRIDE + t0] = lv;
    }
}

// ---------------------------------------------------------------------------
// Kernel 1: standardize each (b,c) image; emit packed {bf16 hi, bf16 lo}.
// ---------------------------------------------------------------------------
__global__ __launch_bounds__(256)
void standardize_kernel(const float* __restrict__ x, unsigned* __restrict__ xp) {
    const int img = blockIdx.x;                    // 0..255 = b*32+c
    const float* __restrict__ src = x + (size_t)img * NPIX;
    unsigned* __restrict__ dst    = xp + (size_t)img * NPIX;
    const int t = threadIdx.x;

    float s1 = 0.f, s2 = 0.f;
    for (int i = t; i < NPIX; i += 256) {
        float v = src[i];
        s1 += v; s2 += v * v;
    }
    __shared__ float r1[256], r2[256];
    r1[t] = s1; r2[t] = s2;
    __syncthreads();
    for (int s = 128; s > 0; s >>= 1) {
        if (t < s) { r1[t] += r1[t + s]; r2[t] += r2[t + s]; }
        __syncthreads();
    }
    const float n     = (float)NPIX;
    const float mean  = r1[0] / n;
    const float var   = fmaxf((r2[0] - n * mean * mean) / (n - 1.0f), 0.f);
    const float sd    = sqrtf(var);
    const float scale = (sd < 1e-9f) ? 0.f : 1.f / (sd * 128.0f);   // sqrt(16384)=128

    for (int i = t; i < NPIX; i += 256) {
        float v = (src[i] - mean) * scale;
        unsigned hb = bf_bits(v);
        unsigned lb = bf_bits(v - bfbits_to_f(hb));
        dst[i] = hb | (lb << 16);
    }
}

// ---------------------------------------------------------------------------
// Sliding-window Gram body, templated on shift count (branch-free hot loop).
// ---------------------------------------------------------------------------
template<int NSH>
__device__ __forceinline__ void xcorr_body(const unsigned* __restrict__ base,
                                           float* __restrict__ out,
                                           __bf16* Aw, __bf16* Bb,
                                           int b, int sy0, int gx, int dx0,
                                           int tid, int lane, int w) {
    const int tp = w / 7;                      // tile position 0..2
    const int q  = w % 7;                      // sy within group
    const int sy = sy0 + q;
    const int Mb = (tp == 2) ? 16 : 0;         // tiles (0,0),(0,16),(16,16)
    const int Nb = (tp == 0) ? 0  : 16;

    v8f acc[NSH];
    #pragma unroll
    for (int s = 0; s < NSH; ++s) acc[s] = (v8f){0.f,0.f,0.f,0.f,0.f,0.f,0.f,0.f};

    // ---- prologue: window rows sy0-10 .. sy0-4, B row 0 ----
    for (int r = 0; r < 7; ++r) {
        const int ri = sy0 + r - 10;
        fill_a_row(Aw + (ri & 7) * AW_SLOT_E, base, ri & 127, dx0, tid);
    }
    fill_b_row(Bb, base, 0, tid);
    __syncthreads();

    for (int y = 0; y < HW; ++y) {
        // ---- fill next: A row y+sy0-3 (slot unused this step), B row y+1 ----
        const int rn = y + sy0 - 3;
        fill_a_row(Aw + (rn & 7) * AW_SLOT_E, base, rn & 127, dx0, tid);
        fill_b_row(Bb + ((y + 1) & 1) * BB_BUF_E, base, (y + 1) & 127, tid);

        // ---- prefetch 2 rows ahead (waves 19/20, global_prefetch_b8) ----
        if (w == 20) {
            const unsigned* pA = base + lane * NPIX + ((y + sy0 - 2) & 127) * HW;
            #pragma unroll
            for (int qq = 0; qq < 4; ++qq) __builtin_prefetch(pA + qq * 32, 0, 1);
        } else if (w == 19) {
            const unsigned* pB = base + lane * NPIX + ((y + 2) & 127) * HW;
            #pragma unroll
            for (int qq = 0; qq < 4; ++qq) __builtin_prefetch(pB + qq * 32, 0, 1);
        }

        // ---- compute step y: A row (y+sy-10), B row y ----
        const __bf16* aw0 = Aw + ((y + sy0 + q - 10) & 7) * AW_SLOT_E;
        const __bf16* aw1 = aw0 + AW_PLANE_E;
        const __bf16* bb0 = Bb + (y & 1) * BB_BUF_E;
        const __bf16* bb1 = bb0 + AW_PLANE_E;
        #pragma unroll
        for (int kc = 0; kc < 4; ++kc) {
            const int kb = kc * 32;
            const v16bf bh = ldB(bb0, Nb, kb, lane);
            const v16bf bl = ldB(bb1, Nb, kb, lane);
            #pragma unroll
            for (int si = 0; si < NSH; ++si) {
                const v16bf ah = ldA(aw0, Mb, kb, si << 3, lane);
                const v16bf al = ldA(aw1, Mb, kb, si << 3, lane);
                acc[si] = wmma_bf16(ah, bh, acc[si]);  // hi*hi
                acc[si] = wmma_bf16(ah, bl, acc[si]);  // hi*lo
                acc[si] = wmma_bf16(al, bh, acc[si]);  // lo*hi
            }
        }
        __syncthreads();       // orders this step's reads before next overwrite
    }

    // ---- write upper triangle; C layout: VGPR v, lane -> (M=v+8*(lane>=16), N=lane&15)
    const int j   = Nb + (lane & 15);
    const int mhi = (lane >> 4) << 3;
    #pragma unroll
    for (int si = 0; si < NSH; ++si) {
        const int sx = gx + (si << 3);
        const v8f c = acc[si];
        #pragma unroll
        for (int v = 0; v < 8; ++v) {
            const int i = Mb + v + mhi;
            if (i <= j) {
                const int p = i * CHANS - (i * (i - 1)) / 2 + (j - i);
                out[(((size_t)b * NPAIR + p) * NS + sy) * NS + sx] = c[v];
            }
        }
    }
}

// ---------------------------------------------------------------------------
// Kernel 2: 21 waves = 3 tiles x 7 sy; uniform dispatch to NSH=3 / NSH=2 body.
// ---------------------------------------------------------------------------
__global__ __launch_bounds__(NTHR)
void xcorr_kernel(const unsigned* __restrict__ xp, float* __restrict__ out) {
    extern __shared__ __bf16 smem[];
    __bf16* Aw = smem;                         // 8 slots x {hi,lo} x 32x144
    __bf16* Bb = smem + AW_ELEMS;              // 2 bufs  x {hi,lo} x 32x144

    const int tid  = threadIdx.x;
    const int lane = tid & 31;
    const int w    = tid >> 5;                 // 0..20

    const int blk = blockIdx.x;                // 8 * 3 * 8
    const int b   = blk / (NSYG * NGX);
    const int rem = blk % (NSYG * NGX);
    const int syg = rem / NGX;
    const int gx  = rem % NGX;
    const int sy0 = syg * 7;
    const int dx0 = gx - 10;

    const unsigned* __restrict__ base = xp + (size_t)b * CHANS * NPIX;

    if (gx < 5) {   // block-uniform: shifts {g, g+8, g+16}
        xcorr_body<3>(base, out, Aw, Bb, b, sy0, gx, dx0, tid, lane, w);
    } else {        // shifts {g, g+8}
        xcorr_body<2>(base, out, Aw, Bb, b, sy0, gx, dx0, tid, lane, w);
    }
}

// ---------------------------------------------------------------------------
extern "C" void kernel_launch(void* const* d_in, const int* in_sizes, int n_in,
                              void* d_out, int out_size, void* d_ws, size_t ws_size,
                              hipStream_t stream) {
    (void)in_sizes; (void)n_in; (void)out_size; (void)ws_size;
    const float* x = (const float*)d_in[0];
    float* out     = (float*)d_out;
    unsigned* xp   = (unsigned*)d_ws;          // 8*32*16384 u32 = 16.8 MB

    // 184 KB dynamic LDS (gfx1250 WGP: 320 KB) -- host-side attr, capture-safe
    (void)hipFuncSetAttribute((const void*)xcorr_kernel,
                              hipFuncAttributeMaxDynamicSharedMemorySize,
                              SMEM_BYTES);

    standardize_kernel<<<dim3(BATCH * CHANS), dim3(256), 0, stream>>>(x, xp);
    xcorr_kernel<<<dim3(BATCH * NSYG * NGX), dim3(NTHR), SMEM_BYTES, stream>>>(xp, out);
}
